// FARGANCore_23398981829053
// MI455X (gfx1250) — matrix operand order repaired and checked by reading, not compiled, not run on hardware
//
#include <hip/hip_runtime.h>
#include <hip/hip_fp16.h>

typedef __attribute__((ext_vector_type(16))) _Float16 v16h;
typedef __attribute__((ext_vector_type(8)))  _Float16 v8h;
typedef __attribute__((ext_vector_type(8)))  float    v8f;

// ---------------------------------------------------------------------------
// Workspace layout (element offsets in f16 halves within d_ws)
// ---------------------------------------------------------------------------
constexpr size_t alH(size_t x){ return (x + 127) & ~(size_t)127; }

// packed f16 weights (row-major N x Kpad, zero-padded)
constexpr size_t OW_FD1  = 0;                                   // 64 x 32
constexpr size_t OW_FCV  = alH(OW_FD1  + 64*32);                // 128 x 192 (im2col)
constexpr size_t OW_FD2  = alH(OW_FCV  + 128*192);              // 320 x 128
constexpr size_t OW_FWC  = alH(OW_FD2  + 320*128);              // 192 x 352 (K=328)
constexpr size_t OW_FWCG = alH(OW_FWC  + 192*352);              // 192 x 192
constexpr size_t OW_GOUT = alH(OW_FWCG + 192*192);              // 16 x 192 (N=4)
constexpr size_t OW_G1IH = alH(OW_GOUT + 16*192);               // 480 x 288 (K=272)
constexpr size_t OW_G1HH = alH(OW_G1IH + (size_t)480*288);      // 480 x 160
constexpr size_t OW_GLU1 = alH(OW_G1HH + (size_t)480*160);      // 160 x 160
constexpr size_t OW_G2IH = alH(OW_GLU1 + 160*160);              // 384 x 256 (K=240)
constexpr size_t OW_G2HH = alH(OW_G2IH + (size_t)384*256);      // 384 x 128
constexpr size_t OW_GLU2 = alH(OW_G2HH + (size_t)384*128);      // 128 x 128
constexpr size_t OW_G3IH = alH(OW_GLU2 + 128*128);              // 384 x 224 (K=208)
constexpr size_t OW_G3HH = alH(OW_G3IH + (size_t)384*224);      // 384 x 128
constexpr size_t OW_GLU3 = alH(OW_G3HH + (size_t)384*128);      // 128 x 128
constexpr size_t OW_SKIP = alH(OW_GLU3 + 128*128);              // 128 x 704 (K=688)
constexpr size_t OW_SKPG = alH(OW_SKIP + (size_t)128*704);      // 128 x 128
constexpr size_t OW_SIG  = alH(OW_SKPG + 128*128);              // 48 x 128 (N=40)
// activations (f16)
constexpr size_t OA_INF  = alH(OW_SIG  + 48*128);               // 128 x 512 x 32 (fd1 input)
constexpr size_t OA_XH   = alH(OA_INF + (size_t)128*512*32);    // 128 x 516 x 64 (fd1 out)
constexpr size_t OA_Y    = alH(OA_XH  + (size_t)128*516*64);    // 64000 x 128 (conv out)
constexpr size_t OA_COND = alH(OA_Y   + (size_t)64000*128);     // 64000 x 320
constexpr size_t OA_RING = alH(OA_COND+ (size_t)64000*320);     // 128 x 256 excitation ring
constexpr size_t OA_XC   = alH(OA_RING+ 128*256);               // 128 x 352 (xcat)
constexpr size_t OA_FS   = alH(OA_XC  + 128*352);               // 128 x 164 fwc_state
constexpr size_t OA_OUT1 = alH(OA_FS  + 128*164);               // 128 x 192 fwc_out
constexpr size_t OA_U    = alH(OA_OUT1+ 128*192);               // 128 x 320 scratch (t0 / r,z / skip_t)
constexpr size_t OA_AX   = alH(OA_U   + 128*320);               // 128 x 288 GRU input / skip out
constexpr size_t OA_SKIN = alH(OA_AX  + 128*288);               // 128 x 704 skip_in
constexpr size_t OA_END  = alH(OA_SKIN+ (size_t)128*704);
// f32 region (byte offset)
constexpr size_t OB_GAIN = OA_END * 2;                          // 128*500*4 floats

// ---------------------------------------------------------------------------
// WMMA fragment helpers (wave32, A/B in contiguous k = 16*(L>>4)+s, same in both; C/D per ISA)
// ---------------------------------------------------------------------------
__device__ __forceinline__ float sigm(float x){ return 1.f / (1.f + __expf(-x)); }

// A operand 16x32 f16: lane L holds row m=L&15; K = kb..kb+15, kb=(L>>4)*16
__device__ __forceinline__ v16h load_fragA(const _Float16* A, int lda, int m0, int k0, int lane){
  const _Float16* p = A + (size_t)(m0 + (lane & 15)) * lda + k0 + ((lane >> 4) << 4);
  v8h lo = *(const v8h*)p;
  v8h hi = *(const v8h*)(p + 8);
  v16h r;
  #pragma unroll
  for (int i = 0; i < 8; ++i){ r[i] = lo[i]; r[8 + i] = hi[i]; }
  return r;
}

// B operand 32x16 f16 (we feed W as N x K rows): lane L holds col n=L&15; K = kb..kb+15, kb=(L>>4)*16
__device__ __forceinline__ v16h load_fragB(const _Float16* W, int ldw, int n0, int k0, int lane){
  const _Float16* p = W + (size_t)(n0 + (lane & 15)) * ldw + k0 + ((lane >> 4) << 4);
  return *(const v16h*)p;
}

// Accumulate D(16x16) += A[m0:,k]·W[n0:,k]^T over K (multiple of 32)
__device__ __forceinline__ v8f mm_acc(v8f acc, const _Float16* A, int lda,
                                      const _Float16* W, int ldw,
                                      int m0, int n0, int K, int lane){
  __builtin_prefetch((const void*)(W + (size_t)(n0 + (lane & 15)) * ldw), 0, 0);
  for (int k0 = 0; k0 < K; k0 += 32){
    v16h a = load_fragA(A, lda, m0, k0, lane);
    v16h b = load_fragB(W, ldw, n0, k0, lane);
    acc = __builtin_amdgcn_wmma_f32_16x16x32_f16(false, a, false, b, (short)0, acc, false, false);
  }
  return acc;
}

// C/D layout: lane L col n=L&15; VGPR r row = m0 + r + 8*(L>>4)
__device__ __forceinline__ void store_tile(_Float16* D, int ld, int m0, int n0, v8f c, int lane){
  int n  = n0 + (lane & 15);
  int mb = m0 + ((lane >> 4) << 3);
  #pragma unroll
  for (int r = 0; r < 8; ++r) D[(size_t)(mb + r) * ld + n] = (_Float16)c[r];
}

__device__ __forceinline__ v8f load_tile(const _Float16* D, int ld, int m0, int n0, int lane){
  int n  = n0 + (lane & 15);
  int mb = m0 + ((lane >> 4) << 3);
  v8f v;
  #pragma unroll
  for (int r = 0; r < 8; ++r) v[r] = (float)D[(size_t)(mb + r) * ld + n];
  return v;
}

// ---------------------------------------------------------------------------
// Phase 0: weight packing (f32 -> zero-padded f16, row-major N x Kp)
// ---------------------------------------------------------------------------
__global__ void pack_pad(const float* __restrict__ src, _Float16* __restrict__ dst,
                         int N, int K, int Np, int Kp){
  int i = blockIdx.x * blockDim.x + threadIdx.x;
  if (i >= Np * Kp) return;
  int n = i / Kp, k = i - n * Kp;
  float v = (n < N && k < K) ? src[(size_t)n * K + k] : 0.f;
  dst[i] = (_Float16)v;
}

// fconv1_w (128,64,3) -> im2col rows (128 x 192), K index = tap*64 + c
__global__ void pack_conv(const float* __restrict__ src, _Float16* __restrict__ dst){
  int i = blockIdx.x * blockDim.x + threadIdx.x;
  if (i >= 128 * 192) return;
  int n = i / 192, k = i - n * 192;
  int tap = k / 64, c = k - tap * 64;
  dst[i] = (_Float16)src[(size_t)n * 192 + c * 3 + tap];
}

// ---------------------------------------------------------------------------
// Phase A: conditioning network
// ---------------------------------------------------------------------------
// fd1 input: rows (b,t) t<502, cols [feat(20) | pembed(12)]
__global__ void build_inf(const float* __restrict__ feat, const int* __restrict__ period,
                          const float* __restrict__ pembed, _Float16* __restrict__ INF){
  size_t i = (size_t)blockIdx.x * blockDim.x + threadIdx.x;
  if (i >= (size_t)128 * 502 * 32) return;
  int c = (int)(i & 31);
  int row = (int)(i >> 5);
  int b = row / 502, t = row - b * 502;
  float v;
  if (c < 20){
    v = feat[((size_t)b * 504 + t + 2) * 20 + c];
  } else {
    int per = period[b * 504 + t + 2];
    per = per < 32 ? 32 : (per > 254 ? 254 : per);
    v = pembed[(per - 32) * 12 + (c - 20)];
  }
  INF[((size_t)b * 512 + t) * 32 + c] = (_Float16)v;
}

// generic batched D = tanh(A @ W^T), tiles of 16x16, one wave per tile
__global__ __launch_bounds__(256) void gemm_tanh(
    const _Float16* __restrict__ A, const _Float16* __restrict__ W, _Float16* __restrict__ D,
    int Mpad, int Mvalid, int NT, int K, int lda, int ldw, int ldo,
    long long aBatch, long long dBatch){
  int lane = threadIdx.x & 31;
  int gw = (blockIdx.x * 256 + threadIdx.x) >> 5;
  int mt = gw / NT, nt = gw - mt * NT;
  if (mt * 16 >= Mpad) return;
  const _Float16* Ab = A + (size_t)blockIdx.y * aBatch;
  _Float16* Db = D + (size_t)blockIdx.y * dBatch;
  v8f acc = {};
  acc = mm_acc(acc, Ab, lda, W, ldw, mt * 16, nt * 16, K, lane);
  int n  = nt * 16 + (lane & 15);
  int mb = mt * 16 + ((lane >> 4) << 3);
  #pragma unroll
  for (int r = 0; r < 8; ++r){
    int row = mb + r;
    if (row < Mvalid) Db[(size_t)row * ldo + n] = (_Float16)tanhf(acc[r]);
  }
}

// gain = clip(0.2 + 0.8*sigmoid(cond80 . cg_w + cg_b), 0.001, 20)
__global__ void gain_k(const _Float16* __restrict__ COND, const float* __restrict__ cgw,
                       const float* __restrict__ cgb, float* __restrict__ GAIN){
  int i = blockIdx.x * blockDim.x + threadIdx.x;
  if (i >= 128 * 500 * 4) return;
  int sub = i & 3, bf = i >> 2;
  const _Float16* c80 = COND + (size_t)bf * 320 + sub * 80;
  float s = cgb[0];
  #pragma unroll 4
  for (int k = 0; k < 80; ++k) s += (float)c80[k] * cgw[k];
  float g = 0.2f + 0.8f * sigm(s);
  g = fminf(fmaxf(g, 0.001f), 20.f);
  GAIN[i] = g;
}

// ---------------------------------------------------------------------------
// Phase B: persistent recurrent scan (one workgroup, 512 threads = 16 wave32)
// ---------------------------------------------------------------------------
template<int NW>
__device__ __forceinline__ void gru_rz(const _Float16* A, int lda, int Kx,
                                       const _Float16* S, int ldS, int Kh,
                                       const _Float16* Wih, const _Float16* Whh,
                                       _Float16* U, int NT, int wid, int lane){
  for (int tt = wid; tt < 8 * NT; tt += NW){
    int mt = tt / NT, nt = tt - mt * NT;
    int m0 = mt * 16, n0 = nt * 16;
    v8f acc = {};
    acc = mm_acc(acc, A, lda, Wih, Kx, m0, n0, Kx, lane);
    acc = mm_acc(acc, S, ldS, Whh, Kh, m0, n0, Kh, lane);
    v8f o;
    #pragma unroll
    for (int r = 0; r < 8; ++r) o[r] = sigm(acc[r]);
    store_tile(U, 320, m0, n0, o, lane);
  }
}

// n-gate + state update; new-state tiles held in VGPRs across a barrier so
// the in-place LDS state update cannot race the GEMM reads of the old state.
template<int NW, int MAXU>
__device__ __forceinline__ void gru_nup(const _Float16* A, int lda, int Kx,
                                        _Float16* S, int Hd,
                                        const _Float16* WihN, const _Float16* WhhN,
                                        const _Float16* U, int wid, int lane){
  const int NT = Hd / 16;
  const int ntot = 8 * NT;
  v8f hold[MAXU];
  #pragma unroll
  for (int u = 0; u < MAXU; ++u){
    int tt = wid + u * NW;
    if (tt < ntot){
      int mt = tt / NT, nt = tt - mt * NT;
      int m0 = mt * 16, n0 = nt * 16;
      v8f ai = {}; ai = mm_acc(ai, A, lda, WihN, Kx, m0, n0, Kx, lane);
      v8f ah = {}; ah = mm_acc(ah, S, Hd, WhhN, Hd, m0, n0, Hd, lane);
      v8f rr = load_tile(U, 320, m0, n0, lane);
      v8f zz = load_tile(U, 320, m0, Hd + n0, lane);
      v8f hh = load_tile(S, Hd, m0, n0, lane);
      v8f o;
      #pragma unroll
      for (int r = 0; r < 8; ++r){
        float nv = tanhf(ai[r] + rr[r] * ah[r]);
        o[r] = (1.f - zz[r]) * nv + zz[r] * hh[r];
      }
      hold[u] = o;
    }
  }
  __syncthreads();
  #pragma unroll
  for (int u = 0; u < MAXU; ++u){
    int tt = wid + u * NW;
    if (tt < ntot){
      int mt = tt / NT, nt = tt - mt * NT;
      store_tile(S, Hd, mt * 16, nt * 16, hold[u], lane);
    }
  }
  __syncthreads();
}

template<int NW>
__device__ __forceinline__ void glu_to(const _Float16* S, int Hd, const _Float16* Wg,
                                       _Float16* D, int ldd, int NT, int wid, int lane){
  for (int tt = wid; tt < 8 * NT; tt += NW){
    int mt = tt / NT, nt = tt - mt * NT;
    int m0 = mt * 16, n0 = nt * 16;
    v8f acc = {}; acc = mm_acc(acc, S, Hd, Wg, Hd, m0, n0, Hd, lane);
    v8f s = load_tile(S, Hd, m0, n0, lane);
    v8f o;
    #pragma unroll
    for (int r = 0; r < 8; ++r) o[r] = s[r] * sigm(acc[r]);
    store_tile(D, ldd, m0, n0, o, lane);
  }
}

__global__ __launch_bounds__(512) void fargan_scan(
    _Float16* __restrict__ Hp, const int* __restrict__ period,
    const float* __restrict__ goutb, const float* __restrict__ GAIN,
    float* __restrict__ out){
  // GRU states resident in LDS (hot, read twice + updated per step)
  __shared__ __align__(32) _Float16 sS1[128 * 160];
  __shared__ __align__(32) _Float16 sS2[128 * 128];
  __shared__ __align__(32) _Float16 sS3[128 * 128];
  __shared__ float sGain[128];
  __shared__ float sInv[128];
  __shared__ int   sPer[128];
  __shared__ float sPg[128 * 4];

  const int tid = threadIdx.x;
  const int lane = tid & 31;
  const int wid = tid >> 5;
  constexpr int NW = 16;

  const _Float16* Wfwc  = Hp + OW_FWC;
  const _Float16* Wfwcg = Hp + OW_FWCG;
  const _Float16* Wgout = Hp + OW_GOUT;
  const _Float16* W1ih  = Hp + OW_G1IH;
  const _Float16* W1hh  = Hp + OW_G1HH;
  const _Float16* Wglu1 = Hp + OW_GLU1;
  const _Float16* W2ih  = Hp + OW_G2IH;
  const _Float16* W2hh  = Hp + OW_G2HH;
  const _Float16* Wglu2 = Hp + OW_GLU2;
  const _Float16* W3ih  = Hp + OW_G3IH;
  const _Float16* W3hh  = Hp + OW_G3HH;
  const _Float16* Wglu3 = Hp + OW_GLU3;
  const _Float16* Wskip = Hp + OW_SKIP;
  const _Float16* Wskpg = Hp + OW_SKPG;
  const _Float16* Wsig  = Hp + OW_SIG;
  const _Float16* COND  = Hp + OA_COND;
  _Float16* RING = Hp + OA_RING;
  _Float16* XC   = Hp + OA_XC;
  _Float16* FS   = Hp + OA_FS;
  _Float16* OUT1 = Hp + OA_OUT1;
  _Float16* U    = Hp + OA_U;
  _Float16* AX   = Hp + OA_AX;
  _Float16* SKIN = Hp + OA_SKIN;

  // zero initial state
  for (int i = tid; i < 128 * 256; i += 512) RING[i] = (_Float16)0.f;
  for (int i = tid; i < 128 * 164; i += 512) FS[i]   = (_Float16)0.f;
  for (int i = tid; i < 128 * 160; i += 512) sS1[i]  = (_Float16)0.f;
  for (int i = tid; i < 128 * 128; i += 512){ sS2[i] = (_Float16)0.f; sS3[i] = (_Float16)0.f; }
  __threadfence(); __syncthreads();

  for (int t = 0; t < 2000; ++t){
    const int frame = t >> 2, sub = t & 3, base = (t * 40) & 255;

    if (tid < 128){
      int per = period[tid * 504 + 3 + frame];
      per = per < 32 ? 32 : (per > 254 ? 254 : per);
      sPer[tid] = per;
      float g = GAIN[(tid * 500 + frame) * 4 + sub];
      sGain[tid] = g;
      sInv[tid]  = 1.f / (1e-5f + g);
    }
    __syncthreads();

    // build xcat = [fwc_state(164) | cond80(80) | pred(44) | prev(40)] (pad->352)
    for (int i = tid; i < 128 * 328; i += 512){
      int b = i / 328, c = i - b * 328;
      _Float16 v;
      if (c < 164) v = FS[b * 164 + c];
      else if (c < 244) v = COND[((size_t)b * 500 + frame) * 320 + sub * 80 + (c - 164)];
      else if (c < 288){
        int j = c - 244;
        int idx = 256 - sPer[b] + j - 2;
        if (idx >= 256) idx -= sPer[b];
        idx = idx < 0 ? 0 : (idx > 255 ? 255 : idx);
        v = (_Float16)((float)RING[b * 256 + ((base + idx) & 255)] * sInv[b]);
      } else {
        int j = c - 288;
        v = (_Float16)((float)RING[b * 256 + ((base + 216 + j) & 255)] * sInv[b]);
      }
      XC[(size_t)b * 352 + c] = v;
    }
    __threadfence(); __syncthreads();

    // (1) U = tanh(XC @ fwc_w^T)  [128x192], K=352
    for (int tt = wid; tt < 8 * 12; tt += NW){
      int mt = tt / 12, nt = tt - mt * 12;
      v8f acc = {}; acc = mm_acc(acc, XC, 352, Wfwc, 352, mt * 16, nt * 16, 352, lane);
      v8f o;
      #pragma unroll
      for (int r = 0; r < 8; ++r) o[r] = tanhf(acc[r]);
      store_tile(U, 320, mt * 16, nt * 16, o, lane);
    }
    __threadfence(); __syncthreads();

    // (2) fwc_out = U * sigmoid(U @ fwc_glu^T) -> OUT1 and SKIN[:,416:608]; FS <- xcat[:,164:328]
    for (int tt = wid; tt < 8 * 12; tt += NW){
      int mt = tt / 12, nt = tt - mt * 12;
      v8f acc = {}; acc = mm_acc(acc, U, 320, Wfwcg, 192, mt * 16, nt * 16, 192, lane);
      v8f s = load_tile(U, 320, mt * 16, nt * 16, lane);
      v8f o;
      #pragma unroll
      for (int r = 0; r < 8; ++r) o[r] = s[r] * sigm(acc[r]);
      store_tile(OUT1, 192, mt * 16, nt * 16, o, lane);
      store_tile(SKIN + 416, 704, mt * 16, nt * 16, o, lane);
    }
    for (int i = tid; i < 128 * 164; i += 512){
      int b = i / 164, c = i - b * 164;
      FS[i] = XC[(size_t)b * 352 + 164 + c];
    }
    __threadfence(); __syncthreads();

    // (3) pg = sigmoid(fwc_out @ gout^T + b) -> LDS
    for (int tt = wid; tt < 8; tt += NW){
      v8f acc = {}; acc = mm_acc(acc, OUT1, 192, Wgout, 192, tt * 16, 0, 192, lane);
      int n = lane & 15;
      int mb = tt * 16 + ((lane >> 4) << 3);
      if (n < 4){
        float bb = goutb[n];
        #pragma unroll
        for (int r = 0; r < 8; ++r) sPg[(mb + r) * 4 + n] = sigm(acc[r] + bb);
      }
    }
    __threadfence(); __syncthreads();

    // (4) x1 = [fwc_out(192) | pg0*fpitch(40) | prev(40)]  (fpitch = xcat[246:286], prev = xcat[288:328])
    for (int i = tid; i < 128 * 272; i += 512){
      int b = i / 272, c = i - b * 272;
      _Float16 v;
      if (c < 192) v = OUT1[b * 192 + c];
      else if (c < 232) v = (_Float16)(sPg[b * 4 + 0] * (float)XC[(size_t)b * 352 + 246 + (c - 192)]);
      else v = XC[(size_t)b * 352 + 288 + (c - 232)];
      AX[(size_t)b * 288 + c] = v;
    }
    __threadfence(); __syncthreads();

    // GRU1 (H=160)
    gru_rz<NW>(AX, 288, 288, sS1, 160, 160, W1ih, W1hh, U, 20, wid, lane);
    __threadfence(); __syncthreads();
    gru_nup<NW, 5>(AX, 288, 288, sS1, 160, W1ih + (size_t)320 * 288, W1hh + (size_t)320 * 160, U, wid, lane);
    glu_to<NW>(sS1, 160, Wglu1, SKIN + 0, 704, 10, wid, lane);
    __threadfence(); __syncthreads();

    // x2 = [g1(160) | pg1*fpitch | prev]
    for (int i = tid; i < 128 * 240; i += 512){
      int b = i / 240, c = i - b * 240;
      _Float16 v;
      if (c < 160) v = SKIN[(size_t)b * 704 + c];
      else if (c < 200) v = (_Float16)(sPg[b * 4 + 1] * (float)XC[(size_t)b * 352 + 246 + (c - 160)]);
      else v = XC[(size_t)b * 352 + 288 + (c - 200)];
      AX[(size_t)b * 288 + c] = v;
    }
    __threadfence(); __syncthreads();

    // GRU2 (H=128)
    gru_rz<NW>(AX, 288, 256, sS2, 128, 128, W2ih, W2hh, U, 16, wid, lane);
    __threadfence(); __syncthreads();
    gru_nup<NW, 4>(AX, 288, 256, sS2, 128, W2ih + (size_t)256 * 256, W2hh + (size_t)256 * 128, U, wid, lane);
    glu_to<NW>(sS2, 128, Wglu2, SKIN + 160, 704, 8, wid, lane);
    __threadfence(); __syncthreads();

    // x3 = [g2(128) | pg2*fpitch | prev]
    for (int i = tid; i < 128 * 208; i += 512){
      int b = i / 208, c = i - b * 208;
      _Float16 v;
      if (c < 128) v = SKIN[(size_t)b * 704 + 160 + c];
      else if (c < 168) v = (_Float16)(sPg[b * 4 + 2] * (float)XC[(size_t)b * 352 + 246 + (c - 128)]);
      else v = XC[(size_t)b * 352 + 288 + (c - 168)];
      AX[(size_t)b * 288 + c] = v;
    }
    __threadfence(); __syncthreads();

    // GRU3 (H=128)
    gru_rz<NW>(AX, 288, 224, sS3, 128, 128, W3ih, W3hh, U, 16, wid, lane);
    __threadfence(); __syncthreads();
    gru_nup<NW, 4>(AX, 288, 224, sS3, 128, W3ih + (size_t)256 * 224, W3hh + (size_t)256 * 128, U, wid, lane);
    glu_to<NW>(sS3, 128, Wglu3, SKIN + 288, 704, 8, wid, lane);
    // SKIN tail: [608:648] = pg3*fpitch, [648:688] = prev
    for (int i = tid; i < 128 * 80; i += 512){
      int b = i / 80, c = i - b * 80;
      _Float16 v;
      if (c < 40) v = (_Float16)(sPg[b * 4 + 3] * (float)XC[(size_t)b * 352 + 246 + c]);
      else v = XC[(size_t)b * 352 + 288 + (c - 40)];
      SKIN[(size_t)b * 704 + 608 + c] = v;
    }
    __threadfence(); __syncthreads();

    // (15) skip_t = tanh(skip_in @ skip_w^T), K=704 -> U[:,0:128]
    for (int tt = wid; tt < 8 * 8; tt += NW){
      int mt = tt >> 3, nt = tt & 7;
      v8f acc = {}; acc = mm_acc(acc, SKIN, 704, Wskip, 704, mt * 16, nt * 16, 704, lane);
      v8f o;
      #pragma unroll
      for (int r = 0; r < 8; ++r) o[r] = tanhf(acc[r]);
      store_tile(U, 320, mt * 16, nt * 16, o, lane);
    }
    __threadfence(); __syncthreads();
    // (16) skip = U * sigmoid(U @ skip_glu^T) -> AX[:,0:128]
    for (int tt = wid; tt < 8 * 8; tt += NW){
      int mt = tt >> 3, nt = tt & 7;
      v8f acc = {}; acc = mm_acc(acc, U, 320, Wskpg, 128, mt * 16, nt * 16, 128, lane);
      v8f s = load_tile(U, 320, mt * 16, nt * 16, lane);
      v8f o;
      #pragma unroll
      for (int r = 0; r < 8; ++r) o[r] = s[r] * sigm(acc[r]);
      store_tile(AX, 288, mt * 16, nt * 16, o, lane);
    }
    __threadfence(); __syncthreads();
    // (17) sig = tanh(skip @ sig_w^T) * gain -> d_out and excitation ring
    for (int tt = wid; tt < 8 * 3; tt += NW){
      int mt = tt / 3, nt = tt - mt * 3;
      v8f acc = {}; acc = mm_acc(acc, AX, 288, Wsig, 128, mt * 16, nt * 16, 128, lane);
      int n  = nt * 16 + (lane & 15);
      int mb = mt * 16 + ((lane >> 4) << 3);
      if (n < 40){
        #pragma unroll
        for (int r = 0; r < 8; ++r){
          int b = mb + r;
          float val = tanhf(acc[r]) * sGain[b];
          out[(size_t)b * 80000 + (size_t)t * 40 + n] = val;
          RING[b * 256 + ((base + n) & 255)] = (_Float16)val;
        }
      }
    }
    __threadfence(); __syncthreads();
  }
}

// ---------------------------------------------------------------------------
// Host launcher
// ---------------------------------------------------------------------------
extern "C" void kernel_launch(void* const* d_in, const int* in_sizes, int n_in,
                              void* d_out, int out_size, void* d_ws, size_t ws_size,
                              hipStream_t stream){
  const float* features   = (const float*)d_in[0];
  const int*   period     = (const int*)  d_in[1];
  const float* pembed     = (const float*)d_in[2];
  const float* fd1_w      = (const float*)d_in[3];
  const float* fconv1_w   = (const float*)d_in[4];
  const float* fd2_w      = (const float*)d_in[5];
  const float* cg_w       = (const float*)d_in[6];
  const float* cg_b       = (const float*)d_in[7];
  const float* fwc_w      = (const float*)d_in[8];
  const float* fwc_glu_w  = (const float*)d_in[9];
  const float* g1_ih      = (const float*)d_in[10];
  const float* g1_hh      = (const float*)d_in[11];
  const float* glu1_w     = (const float*)d_in[12];
  const float* g2_ih      = (const float*)d_in[13];
  const float* g2_hh      = (const float*)d_in[14];
  const float* glu2_w     = (const float*)d_in[15];
  const float* g3_ih      = (const float*)d_in[16];
  const float* g3_hh      = (const float*)d_in[17];
  const float* glu3_w     = (const float*)d_in[18];
  const float* skip_w     = (const float*)d_in[19];
  const float* skip_glu_w = (const float*)d_in[20];
  const float* sig_w      = (const float*)d_in[21];
  const float* gout_w     = (const float*)d_in[22];
  const float* gout_b     = (const float*)d_in[23];

  _Float16* H  = (_Float16*)d_ws;
  float* GAINp = (float*)((char*)d_ws + OB_GAIN);
  float* out   = (float*)d_out;

  // --- pack weights to padded f16 ---
  struct PD { const float* src; size_t off; int N, K, Np, Kp; };
  const PD pd[] = {
    { fd1_w,      OW_FD1,  64, 32,  64, 32  },
    { fd2_w,      OW_FD2,  320,128, 320,128 },
    { fwc_w,      OW_FWC,  192,328, 192,352 },
    { fwc_glu_w,  OW_FWCG, 192,192, 192,192 },
    { gout_w,     OW_GOUT, 4,  192, 16, 192 },
    { g1_ih,      OW_G1IH, 480,272, 480,288 },
    { g1_hh,      OW_G1HH, 480,160, 480,160 },
    { glu1_w,     OW_GLU1, 160,160, 160,160 },
    { g2_ih,      OW_G2IH, 384,240, 384,256 },
    { g2_hh,      OW_G2HH, 384,128, 384,128 },
    { glu2_w,     OW_GLU2, 128,128, 128,128 },
    { g3_ih,      OW_G3IH, 384,208, 384,224 },
    { g3_hh,      OW_G3HH, 384,128, 384,128 },
    { glu3_w,     OW_GLU3, 128,128, 128,128 },
    { skip_w,     OW_SKIP, 128,688, 128,704 },
    { skip_glu_w, OW_SKPG, 128,128, 128,128 },
    { sig_w,      OW_SIG,  40, 128, 48, 128 },
  };
  for (const PD& p : pd){
    int tot = p.Np * p.Kp;
    pack_pad<<<(tot + 255) / 256, 256, 0, stream>>>(p.src, H + p.off, p.N, p.K, p.Np, p.Kp);
  }
  pack_conv<<<(128 * 192 + 255) / 256, 256, 0, stream>>>(fconv1_w, H + OW_FCV);

  // --- phase A: conditioning net ---
  build_inf<<<(int)(((size_t)128 * 502 * 32 + 255) / 256), 256, 0, stream>>>(
      features, period, pembed, H + OA_INF);
  // fd1: per-batch, 512x32 -> 516x64 (tanh), 502 valid rows
  gemm_tanh<<<dim3(16, 128), 256, 0, stream>>>(
      H + OA_INF, H + OW_FD1, H + OA_XH, 512, 502, 4, 32, 32, 32, 64,
      (long long)512 * 32, (long long)516 * 64);
  // conv1 as im2col GEMM: per-batch, rows overlap (lda=64), K=192, 500 valid rows
  gemm_tanh<<<dim3(32, 128), 256, 0, stream>>>(
      H + OA_XH, H + OW_FCV, H + OA_Y, 512, 500, 8, 192, 64, 192, 128,
      (long long)516 * 64, (long long)500 * 128);
  // fd2: 64000x128 -> 64000x320 (tanh)
  gemm_tanh<<<dim3(10000, 1), 256, 0, stream>>>(
      H + OA_Y, H + OW_FD2, H + OA_COND, 64000, 64000, 20, 128, 128, 128, 320, 0, 0);
  // per-substep gains
  gain_k<<<(128 * 500 * 4 + 255) / 256, 256, 0, stream>>>(H + OA_COND, cg_w, cg_b, GAINp);

  // --- phase B: persistent autoregressive scan (single WGP) ---
  fargan_scan<<<1, 512, 0, stream>>>(H, period, gout_b, GAINp, out);
}
